// TransposeAndAttend_49100066128062
// MI455X (gfx1250) — matrix operand order repaired and checked by reading, not compiled, not run on hardware
//
#include <hip/hip_runtime.h>

// Problem constants (match reference)
#define B_   4
#define S_   2048
#define D_   512
#define H_   8
#define KD_  64
#define VD_  64
#define HVD_ 512          // H_*VD_
#define BS_  (B_*S_)

typedef _Float16 v16h __attribute__((ext_vector_type(16)));
typedef _Float16 v8h  __attribute__((ext_vector_type(8)));
typedef _Float16 v4h  __attribute__((ext_vector_type(4)));
typedef float    v8f  __attribute__((ext_vector_type(8)));
typedef unsigned int v4u  __attribute__((ext_vector_type(4)));
typedef int          v8i_ __attribute__((ext_vector_type(8)));
typedef int          v4i_ __attribute__((ext_vector_type(4)));

#if defined(__has_builtin)
# if __has_builtin(__builtin_amdgcn_tensor_load_to_lds)
#  define HAVE_TDM 1
# else
#  define HAVE_TDM 0
# endif
#else
# define HAVE_TDM 0
#endif

union FragA { v16h v; v8h h[2]; };

__device__ __forceinline__ v8f wmma32(v16h a, v16h b, v8f c) {
  // D = A(16x32 f16) * B(32x16 f16) + C(16x16 f32)
  return __builtin_amdgcn_wmma_f32_16x16x32_f16(
      /*neg_a=*/false, a, /*neg_b=*/false, b,
      /*c_mod=*/(short)0, c, /*reuse_a=*/false, /*reuse_b=*/false);
}

// A-fragment (16x32 f16, ISA 7.12.2): lane holds row m=lane%16,
// K = kb + 8*half + {0..7} and kb + 16 + 8*half + {0..7}
__device__ __forceinline__ v16h load_fragA(const _Float16* base, int stride,
                                           int row, int kb, int half) {
  FragA f;
  const _Float16* p = base + (size_t)row * stride + kb + 8 * half;
  f.h[0] = *(const v8h*)(p);
  f.h[1] = *(const v8h*)(p + 16);
  return f.v;
}
// B-fragment (32x16 f16): lane holds col n=lane%16 (B^T row-major "row"),
// K = kb + 8*half + {0..7} and kb + 16 + 8*half + {0..7}, loaded as two 16B chunks
// (works for 16B-aligned LDS rows and any 16B-aligned global row stride).
__device__ __forceinline__ v16h load_fragB16(const _Float16* base, int stride,
                                             int row, int kb, int half) {
  FragA f;
  const _Float16* p = base + (size_t)row * stride + kb + 8 * half;
  f.h[0] = *(const v8h*)(p);
  f.h[1] = *(const v8h*)(p + 16);
  return f.v;
}

#if HAVE_TDM
// 2D tile load via Tensor Data Mover. Descriptor per CDNA5 ISA ch.8.
__device__ __forceinline__ void tdm_load_2d(unsigned int lds_off, const void* gptr,
                                            unsigned int tens_d0, unsigned int tens_d1,
                                            unsigned int tile_d0, unsigned int tile_d1,
                                            unsigned long long stride0,
                                            unsigned int flags_w0) {
  unsigned long long ga = (unsigned long long)(uintptr_t)gptr;
  v4u g0;
  g0.x = 1u;                                           // count=1 (valid D#)
  g0.y = lds_off;                                      // lds_addr (bytes)
  g0.z = (unsigned int)ga;                             // global_addr[31:0]
  g0.w = (unsigned int)((ga >> 32) & 0x01FFFFFFu) | (2u << 30);   // addr[56:32] | type=2
  v8i_ g1;
  g1[0] = (int)flags_w0;
  g1[1] = (int)((tens_d0 & 0xFFFFu) << 16);                        // tensor_dim0[15:0]
  g1[2] = (int)((tens_d0 >> 16) | ((tens_d1 & 0xFFFFu) << 16));    // dim0 hi | dim1 lo
  g1[3] = (int)((tens_d1 >> 16) | (tile_d0 << 16));                // dim1 hi | tile_dim0
  g1[4] = (int)(tile_d1 & 0xFFFFu);                                // tile_dim1 | tile_dim2=0
  g1[5] = (int)(unsigned int)stride0;                              // tensor_dim0_stride lo
  g1[6] = (int)(unsigned int)((stride0 >> 32) & 0xFFFFu);          // stride hi | dim1_stride=0
  g1[7] = 0;
  v4i_ z4 = {0, 0, 0, 0};
#if defined(__clang_major__) && (__clang_major__ >= 23)
  v8i_ z8 = {0, 0, 0, 0, 0, 0, 0, 0};
  __builtin_amdgcn_tensor_load_to_lds(g0, g1, z4, z4, z8, 0);
#else
  __builtin_amdgcn_tensor_load_to_lds(g0, g1, z4, z4, 0);
#endif
}
// pad cfg: 2B data; after every N DWORDs insert 4 DWORDs (8 halfs) of pad
#define TFLG_ROW32H ((1u << 16) | (1u << 20) | (3u << 22) | (3u << 25))  // rows of 32 halfs
#define TFLG_ROW64H ((1u << 16) | (1u << 20) | (4u << 22) | (3u << 25))  // rows of 64 halfs
#endif

// ---------------- elementwise convert f32 -> f16 (x4 vectorized) -------------
__global__ void cvt4_kernel(const float4* __restrict__ in, v4h* __restrict__ out, int n4) {
  int i = blockIdx.x * blockDim.x + threadIdx.x;
  if (i < n4) {
    float4 v = in[i];
    v4h o = { (_Float16)v.x, (_Float16)v.y, (_Float16)v.z, (_Float16)v.w };
    out[i] = o;
  }
}

// -------- transpose+convert: out[b, c, r] = (f16) in[b, r, c] ---------------
__global__ void transpose_to_f16_kernel(const float* __restrict__ in,
                                        _Float16* __restrict__ out,
                                        int rows, int cols) {
  int idx = blockIdx.x * blockDim.x + threadIdx.x;
  int bt  = blockIdx.y;
  if (idx < rows * cols) {
    int r = idx / cols, c = idx % cols;
    out[(size_t)bt * rows * cols + (size_t)c * rows + r] =
        (_Float16)in[(size_t)bt * rows * cols + idx];
  }
}

// ---------------- fused QKV projection --------------------------------------
// Block = 4 waves, 64 X-rows. Per K-step, the 3x(64x32) weight slab of head h
// is staged in LDS (TDM double-buffered), shared by all 4 waves. A-fragments
// are register double-buffered.
#define WST_ 40     // slab row stride: 32 + 8 halfs pad (16B multiple)
__global__ void __launch_bounds__(128)
qkv_proj_kernel(const _Float16* __restrict__ Xh,   // [BS, D]
                const _Float16* __restrict__ WqT,  // [H, 64, D]
                const _Float16* __restrict__ WkT,
                const _Float16* __restrict__ WvT,
                _Float16* __restrict__ Q,
                _Float16* __restrict__ K,
                _Float16* __restrict__ VT) {
  __shared__ __align__(16) _Float16 Ws[3][2][64][WST_];

  const int lane = threadIdx.x & 31;
  const int wave = threadIdx.x >> 5;
  const int m = lane & 15, g = lane >> 4;
  const int h = blockIdx.y;
  const int row0 = blockIdx.x * 64 + wave * 16;   // [0, BS)
  const int b = row0 / S_, s0 = row0 % S_;

  const _Float16* wq = WqT + (size_t)h * KD_ * D_;
  const _Float16* wk = WkT + (size_t)h * KD_ * D_;
  const _Float16* wv = WvT + (size_t)h * KD_ * D_;

  auto stage = [&](int kb, int buf) {
#if HAVE_TDM
    if (threadIdx.x < 32) {
      tdm_load_2d((unsigned int)(uintptr_t)&Ws[0][buf][0][0], wq + kb,
                  D_, 64, 32, 64, D_, TFLG_ROW32H);
      tdm_load_2d((unsigned int)(uintptr_t)&Ws[1][buf][0][0], wk + kb,
                  D_, 64, 32, 64, D_, TFLG_ROW32H);
      tdm_load_2d((unsigned int)(uintptr_t)&Ws[2][buf][0][0], wv + kb,
                  D_, 64, 32, 64, D_, TFLG_ROW32H);
    }
#else
    for (int idx = threadIdx.x; idx < 192; idx += 128) {
      const int mat = idx >> 6, row = idx & 63;
      const _Float16* s = (mat == 0 ? wq : mat == 1 ? wk : wv) + (size_t)row * D_ + kb;
      _Float16* dpp = &Ws[mat][buf][row][0];
#pragma unroll
      for (int c = 0; c < 32; c += 8)
        *(v8h*)(dpp + c) = *(const v8h*)(s + c);
    }
#endif
  };

  v8f aq[4] = {}, ak[4] = {}, av[4] = {};
  stage(0, 0);
  v16h a_cur = load_fragA(Xh, D_, row0 + m, 0, g);

  for (int n = 0; n < D_ / 32; ++n) {
    const int buf = n & 1;
    if (n + 1 < D_ / 32) stage((n + 1) * 32, buf ^ 1);
#if HAVE_TDM
    if (threadIdx.x < 32) {
      if (n + 1 < D_ / 32) __builtin_amdgcn_s_wait_tensorcnt((short)3);
      else                 __builtin_amdgcn_s_wait_tensorcnt((short)0);
    }
#endif
    __syncthreads();

    // prefetch next A-fragment while this step's WMMAs run
    v16h a_nxt = a_cur;
    if (n + 1 < D_ / 32) a_nxt = load_fragA(Xh, D_, row0 + m, (n + 1) * 32, g);

#pragma unroll
    for (int nt = 0; nt < 4; ++nt)
      aq[nt] = wmma32(a_cur, load_fragB16(&Ws[0][buf][0][0], WST_, nt * 16 + m, 0, g), aq[nt]);
#pragma unroll
    for (int nt = 0; nt < 4; ++nt)
      ak[nt] = wmma32(a_cur, load_fragB16(&Ws[1][buf][0][0], WST_, nt * 16 + m, 0, g), ak[nt]);
#pragma unroll
    for (int nt = 0; nt < 4; ++nt)
      av[nt] = wmma32(a_cur, load_fragB16(&Ws[2][buf][0][0], WST_, nt * 16 + m, 0, g), av[nt]);

    a_cur = a_nxt;
    __syncthreads();
  }

  const size_t qb = ((size_t)(b * H_ + h) * S_ + s0) * KD_;
#pragma unroll
  for (int nt = 0; nt < 4; ++nt) {
#pragma unroll
    for (int r = 0; r < 8; ++r) {
      const int mr = r + 8 * g;                  // C/D layout: m = vgpr + 8*half
      Q[qb + (size_t)mr * KD_ + nt * 16 + m] = (_Float16)aq[nt][r];
      K[qb + (size_t)mr * KD_ + nt * 16 + m] = (_Float16)ak[nt][r];
      VT[((size_t)(b * H_ + h) * VD_ + nt * 16 + m) * S_ + s0 + mr] =
          (_Float16)av[nt][r];
    }
  }
}

// ---------------- flash attention ------------------------------------------
// Block = 4 waves (128 thr) covering 64 query rows of one (b,h).
// K/V^T tiles (64x64 f16) staged in LDS by TDM async DMA, double-buffered,
// shared by all 4 waves. Online softmax, no SxS score buffer.
#define KSTR_ 72     // 64 + 8 halfs pad
#define PST_  40     // P-tile row stride (16x32 + 8 pad)
#define NCH_  (S_ / 64)

__global__ void __launch_bounds__(128)
attention_kernel(const _Float16* __restrict__ Q,
                 const _Float16* __restrict__ K,
                 const _Float16* __restrict__ VT,
                 const float* __restrict__ mask,    // [B, S]
                 const float* __restrict__ scales,  // [H]
                 _Float16* __restrict__ ctx) {      // [BS, HVD]
  __shared__ __align__(16) _Float16 Kt[2][64][KSTR_];
  __shared__ __align__(16) _Float16 Vt[2][64][KSTR_];
  __shared__ __align__(16) _Float16 Pb[4][16][PST_];

  const int lane = threadIdx.x & 31;
  const int wave = threadIdx.x >> 5;
  const int m = lane & 15, g = lane >> 4;
  const int h = blockIdx.y;
  const int b = blockIdx.z;
  const int i0 = blockIdx.x * 64 + wave * 16;

  const _Float16* Qb = Q  + (size_t)(b * H_ + h) * S_ * KD_;
  const _Float16* Kb = K  + (size_t)(b * H_ + h) * S_ * KD_;
  const _Float16* Vb = VT + (size_t)(b * H_ + h) * VD_ * S_;
  const float scale = scales[h];

  const v16h aq0 = load_fragA(Qb, KD_, i0 + m, 0,  g);
  const v16h aq1 = load_fragA(Qb, KD_, i0 + m, 32, g);

  _Float16* pbw = &Pb[wave][0][0];

  v8f   acc[4] = {};
  float mrow[8], lrow[8];
#pragma unroll
  for (int r = 0; r < 8; ++r) { mrow[r] = -3.0e38f; lrow[r] = 0.0f; }

  auto stage = [&](int chunk, int buf) {
#if HAVE_TDM
    if (threadIdx.x < 32) {
      tdm_load_2d((unsigned int)(uintptr_t)&Kt[buf][0][0],
                  Kb + (size_t)chunk * 64 * KD_,
                  KD_, 64, KD_, 64, KD_, TFLG_ROW64H);
      tdm_load_2d((unsigned int)(uintptr_t)&Vt[buf][0][0],
                  Vb + (size_t)chunk * 64,
                  S_, 64, 64, 64, S_, TFLG_ROW64H);
    }
#else
    const int t = threadIdx.x;
    const _Float16* src = (t < 64) ? (Kb + (size_t)(chunk * 64 + t) * KD_)
                                   : (Vb + (size_t)(t - 64) * S_ + chunk * 64);
    _Float16* dst = (t < 64) ? &Kt[buf][t][0] : &Vt[buf][t - 64][0];
#pragma unroll
    for (int c = 0; c < 64; c += 8)
      *(v8h*)(dst + c) = *(const v8h*)(src + c);
#endif
  };

  stage(0, 0);
  for (int n = 0; n < NCH_; ++n) {
    const int buf = n & 1;
    if (n + 1 < NCH_) stage(n + 1, buf ^ 1);
#if HAVE_TDM
    if (threadIdx.x < 32) {
      if (n + 1 < NCH_) __builtin_amdgcn_s_wait_tensorcnt((short)2);
      else              __builtin_amdgcn_s_wait_tensorcnt((short)0);
    }
#endif
    __syncthreads();

    const int j0 = n * 64;
#pragma unroll
    for (int sj = 0; sj < 2; ++sj) {            // two 32-key substeps
      const int c0 = sj * 32;
      v8f cst[2];
#pragma unroll
      for (int ct = 0; ct < 2; ++ct) {
        const int rowb = c0 + ct * 16 + m;
        v8f c = {};
        c = wmma32(aq0, load_fragB16(&Kt[buf][0][0], KSTR_, rowb, 0,  g), c);
        c = wmma32(aq1, load_fragB16(&Kt[buf][0][0], KSTR_, rowb, 32, g), c);
        cst[ct] = c;
      }
      const float mj0 = mask[b * S_ + j0 + c0 + m];
      const float mj1 = mask[b * S_ + j0 + c0 + 16 + m];
      float resc[8];
#pragma unroll
      for (int r = 0; r < 8; ++r) {
        float v0 = cst[0][r] * scale, v1 = cst[1][r] * scale;
        float mt = fmaxf(v0, v1);
        mt = fmaxf(mt, __shfl_xor(mt, 1, 32));
        mt = fmaxf(mt, __shfl_xor(mt, 2, 32));
        mt = fmaxf(mt, __shfl_xor(mt, 4, 32));
        mt = fmaxf(mt, __shfl_xor(mt, 8, 32));
        const float mn = fmaxf(mrow[r], mt);
        const float e0 = __expf(v0 - mn);
        const float e1 = __expf(v1 - mn);
        float ts = e0 + e1;
        ts += __shfl_xor(ts, 1, 32);
        ts += __shfl_xor(ts, 2, 32);
        ts += __shfl_xor(ts, 4, 32);
        ts += __shfl_xor(ts, 8, 32);
        const float rs = __expf(mrow[r] - mn);
        lrow[r] = lrow[r] * rs + ts;
        mrow[r] = mn;
        resc[r] = rs;
        pbw[(r + 8 * g) * PST_ + m]      = (_Float16)(e0 * mj0);
        pbw[(r + 8 * g) * PST_ + 16 + m] = (_Float16)(e1 * mj1);
      }
#pragma unroll
      for (int nt = 0; nt < 4; ++nt)
#pragma unroll
        for (int r = 0; r < 8; ++r) acc[nt][r] *= resc[r];
      const v16h pa = load_fragA(pbw, PST_, m, 0, g);
#pragma unroll
      for (int nt = 0; nt < 4; ++nt)
        acc[nt] = wmma32(pa, load_fragB16(&Vt[buf][0][0], KSTR_, nt * 16 + m, c0, g),
                         acc[nt]);
    }
    __syncthreads();
  }

#pragma unroll
  for (int nt = 0; nt < 4; ++nt) {
#pragma unroll
    for (int r = 0; r < 8; ++r) {
      const int i = i0 + r + 8 * g;
      const float mi = mask[b * S_ + i];
      ctx[((size_t)b * S_ + i) * HVD_ + h * VD_ + nt * 16 + m] =
          (_Float16)(acc[nt][r] / lrow[r] * mi);
    }
  }
}

// ---------------- output projection: out = ctx @ Wo -------------------------
// Block = 4 waves, 64 rows; Wo^T slab (64x32) TDM double-buffered in LDS.
__global__ void __launch_bounds__(128)
out_proj_kernel(const _Float16* __restrict__ ctx,  // [BS, HVD]
                const _Float16* __restrict__ WoT,  // [64, HVD]
                float* __restrict__ out) {         // [BS, 64] f32
  __shared__ __align__(16) _Float16 Wo_s[2][64][WST_];

  const int lane = threadIdx.x & 31;
  const int wave = threadIdx.x >> 5;
  const int m = lane & 15, g = lane >> 4;
  const int r0 = blockIdx.x * 64 + wave * 16;

  auto stage = [&](int kb, int buf) {
#if HAVE_TDM
    if (threadIdx.x < 32)
      tdm_load_2d((unsigned int)(uintptr_t)&Wo_s[buf][0][0], WoT + kb,
                  HVD_, 64, 32, 64, HVD_, TFLG_ROW32H);
#else
    const int t = threadIdx.x;
    const int row = t >> 1, hh = t & 1;        // 128 thr -> 64 rows x 2 halves
    const _Float16* s = WoT + (size_t)row * HVD_ + kb + hh * 16;
    _Float16* dpp = &Wo_s[buf][row][hh * 16];
    *(v8h*)(dpp)     = *(const v8h*)(s);
    *(v8h*)(dpp + 8) = *(const v8h*)(s + 8);
#endif
  };

  v8f acc[4] = {};
  stage(0, 0);
  v16h a_cur = load_fragA(ctx, HVD_, r0 + m, 0, g);

  for (int n = 0; n < HVD_ / 32; ++n) {
    const int buf = n & 1;
    if (n + 1 < HVD_ / 32) stage((n + 1) * 32, buf ^ 1);
#if HAVE_TDM
    if (threadIdx.x < 32) {
      if (n + 1 < HVD_ / 32) __builtin_amdgcn_s_wait_tensorcnt((short)1);
      else                   __builtin_amdgcn_s_wait_tensorcnt((short)0);
    }
#endif
    __syncthreads();

    v16h a_nxt = a_cur;
    if (n + 1 < HVD_ / 32) a_nxt = load_fragA(ctx, HVD_, r0 + m, (n + 1) * 32, g);

#pragma unroll
    for (int nt = 0; nt < 4; ++nt)
      acc[nt] = wmma32(a_cur, load_fragB16(&Wo_s[buf][0][0], WST_, nt * 16 + m, 0, g), acc[nt]);

    a_cur = a_nxt;
    __syncthreads();
  }

#pragma unroll
  for (int nt = 0; nt < 4; ++nt)
#pragma unroll
    for (int r = 0; r < 8; ++r)
      out[(size_t)(r0 + r + 8 * g) * VD_ + nt * 16 + m] = acc[nt][r];
}

// ----------------------------------------------------------------------------
extern "C" void kernel_launch(void* const* d_in, const int* in_sizes, int n_in,
                              void* d_out, int out_size, void* d_ws, size_t ws_size,
                              hipStream_t stream) {
  const float* x      = (const float*)d_in[0];   // [B,S,D]
  const float* mask   = (const float*)d_in[1];   // [B,S]
  const float* Wq     = (const float*)d_in[2];   // [H,D,64]
  const float* Wk     = (const float*)d_in[3];
  const float* Wv     = (const float*)d_in[4];
  const float* Wo     = (const float*)d_in[5];   // [512,64]
  const float* scales = (const float*)d_in[6];   // [H]
  float* out = (float*)d_out;
  (void)in_sizes; (void)n_in; (void)out_size; (void)ws_size;

  char* ws = (char*)d_ws;
  size_t off = 0;
  auto alloc = [&](size_t bytes) -> void* {
    void* p = ws + off;
    off = (off + bytes + 255) & ~(size_t)255;
    return p;
  };
  _Float16* Xh  = (_Float16*)alloc((size_t)BS_ * D_ * 2);
  _Float16* WqT = (_Float16*)alloc((size_t)H_ * KD_ * D_ * 2);
  _Float16* WkT = (_Float16*)alloc((size_t)H_ * KD_ * D_ * 2);
  _Float16* WvT = (_Float16*)alloc((size_t)H_ * KD_ * D_ * 2);
  _Float16* WoT = (_Float16*)alloc((size_t)VD_ * HVD_ * 2);
  _Float16* Qh  = (_Float16*)alloc((size_t)B_ * H_ * S_ * KD_ * 2);
  _Float16* Kh  = (_Float16*)alloc((size_t)B_ * H_ * S_ * KD_ * 2);
  _Float16* VTh = (_Float16*)alloc((size_t)B_ * H_ * VD_ * S_ * 2);
  _Float16* Ch  = (_Float16*)alloc((size_t)BS_ * HVD_ * 2);

  {
    int n4 = BS_ * D_ / 4;
    cvt4_kernel<<<(n4 + 255) / 256, 256, 0, stream>>>((const float4*)x, (v4h*)Xh, n4);
  }
  transpose_to_f16_kernel<<<dim3((D_ * KD_ + 255) / 256, H_), 256, 0, stream>>>(Wq, WqT, D_, KD_);
  transpose_to_f16_kernel<<<dim3((D_ * KD_ + 255) / 256, H_), 256, 0, stream>>>(Wk, WkT, D_, KD_);
  transpose_to_f16_kernel<<<dim3((D_ * KD_ + 255) / 256, H_), 256, 0, stream>>>(Wv, WvT, D_, KD_);
  transpose_to_f16_kernel<<<dim3((HVD_ * VD_ + 255) / 256, 1), 256, 0, stream>>>(Wo, WoT, HVD_, VD_);

  // QKV projection: 4 waves/block, TDM-staged weight slabs, double-buffered
  qkv_proj_kernel<<<dim3(BS_ / 64, H_), 128, 0, stream>>>(Xh, WqT, WkT, WvT, Qh, Kh, VTh);

  // Flash attention: 4 waves/block, TDM-staged K/V tiles, double-buffered
  attention_kernel<<<dim3(S_ / 64, H_, B_), 128, 0, stream>>>(Qh, Kh, VTh, mask, scales, Ch);

  // Output projection: 4 waves/block, TDM-staged Wo^T slab
  out_proj_kernel<<<BS_ / 64, 128, 0, stream>>>(Ch, WoT, out);
}
